// FunnelRelMultiheadAttention_24739011625074
// MI455X (gfx1250) — matrix operand order repaired and checked by reading, not compiled, not run on hardware
//
#include <hip/hip_runtime.h>

#define B_   2
#define S_   1024
#define DM   768
#define NH   12
#define DH   64
#define M2   2047          // 2*S - 1
#define SCALEF 0.125f      // 1/sqrt(64)

typedef __attribute__((ext_vector_type(16))) __bf16 v16bf;
typedef __attribute__((ext_vector_type(8)))  __bf16 v8bf;
typedef __attribute__((ext_vector_type(8)))  float  v8f;

static __device__ __forceinline__ v8f wmma_bf16(v16bf a, v16bf b, v8f c) {
  // D = A(16x32 bf16) * B(32x16 bf16) + C(16x16 f32)
  return __builtin_amdgcn_wmma_f32_16x16x32_bf16(false, a, false, b, (short)0, c,
                                                 false, false);
}

// A-fragment (16x32 bf16), row-major source: lane row = lane&15.
// Per-lane elements are two contiguous 8-element runs: K = [8h..8h+7] and
// [16+8h..23+8h].  p must point at rowBase + kstep*32 + 8*half.
static __device__ __forceinline__ v16bf ld_afrag(const __bf16* p) {
  v8bf lo = *(const v8bf*)(p);
  v8bf hi = *(const v8bf*)(p + 16);
  return __builtin_shufflevector(lo, hi, 0, 1, 2, 3, 4, 5, 6, 7,
                                 8, 9, 10, 11, 12, 13, 14, 15);
}

// B-fragment (32x16 bf16) where B is a transposed view of a row-major matrix
// (per-lane K runs [8h..8h+7] and [16+8h..23+8h]).  p = rowBase + kstep*32 + 8*half.
static __device__ __forceinline__ v16bf ld_bfrag_rowK(const __bf16* p) {
  v8bf lo = *(const v8bf*)(p);
  v8bf hi = *(const v8bf*)(p + 16);
  return __builtin_shufflevector(lo, hi, 0, 1, 2, 3, 4, 5, 6, 7,
                                 8, 9, 10, 11, 12, 13, 14, 15);
}

// ---------------------------------------------------------------------------
// Kernel 0a: f32 -> bf16 copy with zero-padded tail (n_src <= n_dst).
// ---------------------------------------------------------------------------
__global__ __launch_bounds__(256) void cvt_bf16_kernel(
    const float* __restrict__ src, __bf16* __restrict__ dst, int n_src,
    int n_dst) {
  int i = blockIdx.x * blockDim.x + threadIdx.x;
  if (i >= n_dst) return;
  dst[i] = (i < n_src) ? (__bf16)src[i] : (__bf16)0.0f;
}

// ---------------------------------------------------------------------------
// Kernel 0b: weight prep.  dst[batch][c][r] = (bf16)src[batch][r][c].
// Turns every GEMM B-operand into contiguous-per-lane bf16 (b128 loads).
// ---------------------------------------------------------------------------
__global__ __launch_bounds__(256) void transpose_bf16_kernel(
    const float* __restrict__ src, __bf16* __restrict__ dst, int rows,
    int cols) {
  size_t base = (size_t)blockIdx.y * rows * cols;
  int idx = blockIdx.x * blockDim.x + threadIdx.x;
  if (idx >= rows * cols) return;
  int r = idx / cols, c = idx % cols;
  dst[base + (size_t)c * rows + r] = (__bf16)src[base + idx];
}

// ---------------------------------------------------------------------------
// Kernel 1: grouped q/k/v projections.  One wave per 16x16 output tile.
// p=0 -> q (emits q+u and q+v, both *SCALE), p=1 -> k (+bk),
// p=2 -> v (+bv), stored TRANSPOSED per head: vT[b][n][dh][s].
// ---------------------------------------------------------------------------
__global__ __launch_bounds__(256) void proj_qkv_kernel(
    const __bf16* __restrict__ qbf, const __bf16* __restrict__ WqT,
    const __bf16* __restrict__ WkT, const float* __restrict__ bk,
    const __bf16* __restrict__ WvT, const float* __restrict__ bv,
    const float* __restrict__ rwb, const float* __restrict__ rrb,
    __bf16* __restrict__ qu, __bf16* __restrict__ qv,
    __bf16* __restrict__ ko, __bf16* __restrict__ vT) {
  int wid = blockIdx.x * (blockDim.x >> 5) + (threadIdx.x >> 5);
  int lane = threadIdx.x & 31;
  int half = lane >> 4, l16 = lane & 15;
  int ct = wid % 48;
  int mt = (wid / 48) % 128;
  int p = wid / (48 * 128);
  if (p >= 3) return;
  int row0 = mt * 16;  // row = b*S + s
  int c0 = ct * 16;    // output channel base (tiles never straddle groups)
  int g = c0 / 192;
  const __bf16* WT = (p == 0) ? WqT : (p == 1) ? WkT : WvT;  // [g][o][i]
  const __bf16* abase =
      qbf + (size_t)(row0 + l16) * DM + g * 192 + half * 8;
  const __bf16* bptr = WT + (size_t)g * 192 * 192 +
                       (size_t)((c0 % 192) + l16) * 192 + half * 8;
  v8f acc = {};
#pragma unroll
  for (int ks = 0; ks < 6; ++ks) {  // K = 192
    v16bf a = ld_afrag(abase + ks * 32);
    v16bf b = ld_bfrag_rowK(bptr + ks * 32);
    acc = wmma_bf16(a, b, acc);
  }
  int cc = c0 + l16;
  if (p == 2) {  // transposed per-head V store
    int n = cc / DH, hh = cc % DH;
    int bb = row0 >> 10;  // 16-row tiles never straddle the batch boundary
    int s0 = row0 & (S_ - 1);
    size_t vbase = ((size_t)(bb * NH + n) * DH + hh) * S_;
    float bias = bv[cc];
#pragma unroll
    for (int r = 0; r < 8; ++r)
      vT[vbase + s0 + r + half * 8] = (__bf16)(acc[r] + bias);
  } else if (p == 1) {
    float bias = bk[cc];
#pragma unroll
    for (int r = 0; r < 8; ++r)
      ko[(size_t)(row0 + r + half * 8) * DM + cc] = (__bf16)(acc[r] + bias);
  } else {
    float bu = rwb[cc], br = rrb[cc];
#pragma unroll
    for (int r = 0; r < 8; ++r) {
      size_t o = (size_t)(row0 + r + half * 8) * DM + cc;
      qu[o] = (__bf16)((acc[r] + bu) * SCALEF);
      qv[o] = (__bf16)((acc[r] + br) * SCALEF);
    }
  }
}

// ---------------------------------------------------------------------------
// Kernel 2: r_head[t, n*64+h] = position_embeds[t,:] @ r_kernel[:, n, h].
// pe_bf row 2047 is pre-zeroed, so row 2047 of r_head comes out zero (pad
// for the rel-shift band) with no guards in the hot loop.
// ---------------------------------------------------------------------------
__global__ __launch_bounds__(256) void rhead_kernel(
    const __bf16* __restrict__ pebf, const __bf16* __restrict__ rkT,
    __bf16* __restrict__ rh) {
  int wid = blockIdx.x * (blockDim.x >> 5) + (threadIdx.x >> 5);
  int lane = threadIdx.x & 31, half = lane >> 4, l16 = lane & 15;
  int ct = wid % 48, mt = wid / 48;
  if (mt >= 128) return;
  const __bf16* abase = pebf + (size_t)(mt * 16 + l16) * DM + half * 8;
  int c0 = ct * 16;
  const __bf16* bptr = rkT + (size_t)(c0 + l16) * DM + half * 8;
  v8f acc = {};
#pragma unroll
  for (int ks = 0; ks < 24; ++ks) {  // K = 768
    v16bf a = ld_afrag(abase + ks * 32);
    v16bf b = ld_bfrag_rowK(bptr + ks * 32);
    acc = wmma_bf16(a, b, acc);
  }
#pragma unroll
  for (int r = 0; r < 8; ++r) {
    int tt = mt * 16 + r + half * 8;
    rh[(size_t)tt * DM + c0 + l16] = (__bf16)acc[r];
  }
}

// ---------------------------------------------------------------------------
// Kernel 3: attention.  One wave per (b, head, 16-query tile).
// Phase 1: pos[i][t] = qv[i].r_head[t] -> LDS strip [16][2047] (odd stride ->
//          the diagonal gather t = S-i+j is LDS-bank conflict free).
// Phase 2: streaming softmax; score tiles computed TRANSPOSED (K rows x Q cols)
//          so the probabilities land exactly in the PV WMMA A-fragment slots.
// ---------------------------------------------------------------------------
__global__ __launch_bounds__(32) void attn_kernel(
    const __bf16* __restrict__ qu, const __bf16* __restrict__ qv,
    const __bf16* __restrict__ kb, const __bf16* __restrict__ vT,
    const __bf16* __restrict__ rh, __bf16* __restrict__ av) {
  __shared__ __bf16 pos[16 * M2];  // 65,504 B (fits one 64KB LDS alloc)
  int lane = threadIdx.x & 31, half = lane >> 4, l16 = lane & 15;
  int blk = blockIdx.x;
  int qt = blk & 63;             // 64 query tiles of 16
  int n = (blk >> 6) % NH;       // head
  int b = blk / (64 * NH);       // batch
  int i_base = qt * 16;
  size_t bs = (size_t)b * S_ * DM;
  int hco = n * DH;

  // Query B-fragments (col = query, per-lane K run contiguous): K = 64 -> 2.
  v16bf bqu[2], bqv[2];
  {
    const __bf16* qup = qu + bs + (size_t)(i_base + l16) * DM + hco + half * 8;
    const __bf16* qvp = qv + bs + (size_t)(i_base + l16) * DM + hco + half * 8;
#pragma unroll
    for (int ks = 0; ks < 2; ++ks) {
      bqu[ks] = ld_bfrag_rowK(qup + ks * 32);
      bqv[ks] = ld_bfrag_rowK(qvp + ks * 32);
    }
  }

  // ---- Phase 1: pos^T tiles (r_head rows x qv cols) into LDS ----
  for (int tt = 0; tt < 128; ++tt) {
    const __bf16* rp = rh + (size_t)(tt * 16 + l16) * DM + hco + half * 8;
    v16bf a0 = ld_afrag(rp);
    v16bf a1 = ld_afrag(rp + 32);
    v8f c = {};
    c = wmma_bf16(a0, bqv[0], c);
    c = wmma_bf16(a1, bqv[1], c);
#pragma unroll
    for (int r = 0; r < 8; ++r) {
      int t = tt * 16 + r + half * 8;
      if (t < M2) pos[l16 * M2 + t] = (__bf16)c[r];  // row = this lane's query
    }
  }
  asm volatile("s_wait_dscnt 0" ::: "memory");

  // ---- Phase 2: stream over keys, 32 per iteration ----
  float mrun = -1e30f, lrun = 0.0f;
  v8f O[4] = {};
  int iq = i_base + l16;
  const __bf16* vhead = vT + (size_t)(b * NH + n) * DH * S_;
  for (int jp = 0; jp < 32; ++jp) {
    int j0 = jp * 32;
    float s[16];
#pragma unroll
    for (int sub = 0; sub < 2; ++sub) {
      int jb = j0 + sub * 16;
      const __bf16* kp = kb + bs + (size_t)(jb + l16) * DM + hco + half * 8;
      __builtin_prefetch(kp + 32 * DM, 0, 3);  // next key tile
      v16bf a0 = ld_afrag(kp);
      v16bf a1 = ld_afrag(kp + 32);
      v8f c = {};
      c = wmma_bf16(a0, bqu[0], c);  // S^T[j][i]: lane col = query iq
      c = wmma_bf16(a1, bqu[1], c);
#pragma unroll
      for (int r = 0; r < 8; ++r) {
        int j = jb + r + half * 8;
        // HF _relative_shift_gather: F = i*(2S-2)+j+S ; (i0,t0)=(F/M2, F%M2)
        int F = iq * (2 * S_ - 2) + j + S_;
        int i0 = F / M2;
        int t0 = F - i0 * M2;
        s[sub * 8 + r] = c[r] + (float)pos[(i0 - i_base) * M2 + t0];
      }
    }
    // Per-lane stats are per-query; merge the two half-wave lanes only.
    float tmax = s[0];
#pragma unroll
    for (int e = 1; e < 16; ++e) tmax = fmaxf(tmax, s[e]);
    tmax = fmaxf(tmax, __shfl_xor(tmax, 16, 32));
    float mnew = fmaxf(mrun, tmax);
    float alpha = __expf(mrun - mnew);
    mrun = mnew;
    v16bf pa;  // P lands directly in the PV A-fragment layout
    float ls = 0.0f;
#pragma unroll
    for (int e = 0; e < 16; ++e) {
      float pexp = __expf(s[e] - mnew);
      ls += pexp;
      pa[e] = (__bf16)pexp;
    }
    lrun = lrun * alpha + ls;
    // Rescale accumulators: O rows are queries r+8*half -> fetch their alpha.
    float af[8];
#pragma unroll
    for (int r = 0; r < 8; ++r) af[r] = __shfl(alpha, r + half * 8, 32);
#pragma unroll
    for (int c4 = 0; c4 < 4; ++c4)
#pragma unroll
      for (int r = 0; r < 8; ++r) O[c4][r] *= af[r];
    // PV: B = V^T rows are head-dims -> per-lane K runs over keys in table order.
#pragma unroll
    for (int c4 = 0; c4 < 4; ++c4) {
      const __bf16* vp =
          vhead + (size_t)(c4 * 16 + l16) * S_ + j0 + half * 8;
      v16bf vfrag = ld_bfrag_rowK(vp);
      O[c4] = wmma_bf16(pa, vfrag, O[c4]);
    }
  }
  float ltot = lrun + __shfl_xor(lrun, 16, 32);
  float linv = 1.0f / ltot;
  float lr[8];
#pragma unroll
  for (int r = 0; r < 8; ++r) lr[r] = __shfl(linv, r + half * 8, 32);
#pragma unroll
  for (int c4 = 0; c4 < 4; ++c4)
#pragma unroll
    for (int r = 0; r < 8; ++r) {
      int row = i_base + r + half * 8;
      av[bs + (size_t)row * DM + hco + c4 * 16 + l16] =
          (__bf16)(O[c4][r] * lr[r]);
    }
}

// ---------------------------------------------------------------------------
// Kernel 4: h = query + attn_vec @ Wo + bo
// ---------------------------------------------------------------------------
__global__ __launch_bounds__(256) void out_proj_kernel(
    const __bf16* __restrict__ av, const __bf16* __restrict__ WoT,
    const float* __restrict__ bo, const float* __restrict__ qin,
    float* __restrict__ h) {
  int wid = blockIdx.x * (blockDim.x >> 5) + (threadIdx.x >> 5);
  int lane = threadIdx.x & 31, half = lane >> 4, l16 = lane & 15;
  int ct = wid % 48, mt = wid / 48;
  if (mt >= 128) return;
  int row0 = mt * 16, c0 = ct * 16;
  const __bf16* ap = av + (size_t)(row0 + l16) * DM + half * 8;
  const __bf16* bptr = WoT + (size_t)(c0 + l16) * DM + half * 8;
  v8f acc = {};
#pragma unroll
  for (int ks = 0; ks < 24; ++ks) {
    v16bf a = ld_afrag(ap + ks * 32);
    v16bf b = ld_bfrag_rowK(bptr + ks * 32);
    acc = wmma_bf16(a, b, acc);
  }
  int cc = c0 + l16;
#pragma unroll
  for (int r = 0; r < 8; ++r) {
    int row = row0 + r + half * 8;
    h[(size_t)row * DM + cc] = acc[r] + bo[cc] + qin[(size_t)row * DM + cc];
  }
}

// ---------------------------------------------------------------------------
// Kernel 5: row LayerNorm (eps = 1e-9), one block per token.
// ---------------------------------------------------------------------------
__global__ __launch_bounds__(256) void ln_kernel(const float* __restrict__ h,
                                                 const float* __restrict__ gmm,
                                                 const float* __restrict__ bta,
                                                 float* __restrict__ out) {
  int row = blockIdx.x;
  const float* hr = h + (size_t)row * DM;
  int t = threadIdx.x;
  float x[3], s = 0.0f, sq = 0.0f;
#pragma unroll
  for (int k = 0; k < 3; ++k) {
    float v = hr[t + 256 * k];
    x[k] = v;
    s += v;
    sq += v * v;
  }
#pragma unroll
  for (int m = 16; m >= 1; m >>= 1) {
    s += __shfl_xor(s, m, 32);
    sq += __shfl_xor(sq, m, 32);
  }
  __shared__ float ss[8], s2[8], stat[2];
  int w = t >> 5, lane = t & 31;
  if (lane == 0) { ss[w] = s; s2[w] = sq; }
  __syncthreads();
  if (t == 0) {
    float S1 = 0.0f, S2 = 0.0f;
    for (int i = 0; i < 8; ++i) { S1 += ss[i]; S2 += s2[i]; }
    float mu = S1 * (1.0f / DM);
    float var = S2 * (1.0f / DM) - mu * mu;
    stat[0] = mu;
    stat[1] = rsqrtf(var + 1e-9f);
  }
  __syncthreads();
  float mu = stat[0], rs = stat[1];
#pragma unroll
  for (int k = 0; k < 3; ++k) {
    int c = t + 256 * k;
    out[(size_t)row * DM + c] = (x[k] - mu) * rs * gmm[c] + bta[c];
  }
}

// ---------------------------------------------------------------------------
extern "C" void kernel_launch(void* const* d_in, const int* in_sizes, int n_in,
                              void* d_out, int out_size, void* d_ws,
                              size_t ws_size, hipStream_t stream) {
  const float* qin = (const float*)d_in[0];
  const float* pe = (const float*)d_in[1];
  const float* Wq = (const float*)d_in[2];
  const float* Wk = (const float*)d_in[3];
  const float* bk = (const float*)d_in[4];
  const float* Wv = (const float*)d_in[5];
  const float* bv = (const float*)d_in[6];
  const float* rk = (const float*)d_in[7];
  const float* rwb = (const float*)d_in[8];
  const float* rrb = (const float*)d_in[9];
  const float* Wo = (const float*)d_in[10];
  const float* bo = (const float*)d_in[11];
  const float* gmm = (const float*)d_in[12];
  const float* bta = (const float*)d_in[13];
  (void)in_sizes; (void)n_in; (void)out_size; (void)ws_size;

  char* ws = (char*)d_ws;
  const size_t NTOK = (size_t)B_ * S_;                // 2048 tokens
  const size_t bf_mat = NTOK * DM * sizeof(__bf16);   // 3 MB each
  size_t off = 0;
  auto take = [&](size_t bytes) -> void* {
    void* p = ws + off;
    off += (bytes + 255) & ~(size_t)255;
    return p;
  };
  __bf16* qu = (__bf16*)take(bf_mat);
  __bf16* qv = (__bf16*)take(bf_mat);
  __bf16* kbuf = (__bf16*)take(bf_mat);
  __bf16* vT = (__bf16*)take(bf_mat);  // [b][head][dh][s]
  __bf16* rh = (__bf16*)take((size_t)2048 * DM * sizeof(__bf16));
  __bf16* av = (__bf16*)take(bf_mat);
  float* h = (float*)take(NTOK * DM * sizeof(float));
  __bf16* qbf = (__bf16*)take(bf_mat);                         // bf16 query
  __bf16* pebf = (__bf16*)take((size_t)2048 * DM * sizeof(__bf16));
  __bf16* WqT = (__bf16*)take((size_t)4 * 192 * 192 * sizeof(__bf16));
  __bf16* WkT = (__bf16*)take((size_t)4 * 192 * 192 * sizeof(__bf16));
  __bf16* WvT = (__bf16*)take((size_t)4 * 192 * 192 * sizeof(__bf16));
  __bf16* rkT = (__bf16*)take((size_t)DM * DM * sizeof(__bf16));
  __bf16* WoT = (__bf16*)take((size_t)DM * DM * sizeof(__bf16));

  // ---- input/weight prep (tiny) ----
  const int nq = (int)(NTOK * DM);          // 1,572,864
  const int npe_src = M2 * DM;              // 1,572,096
  const int npe_dst = 2048 * DM;            // padded with a zero row
  cvt_bf16_kernel<<<6144, 256, 0, stream>>>(qin, qbf, nq, nq);
  cvt_bf16_kernel<<<6144, 256, 0, stream>>>(pe, pebf, npe_src, npe_dst);
  transpose_bf16_kernel<<<dim3(144, 4), 256, 0, stream>>>(Wq, WqT, 192, 192);
  transpose_bf16_kernel<<<dim3(144, 4), 256, 0, stream>>>(Wk, WkT, 192, 192);
  transpose_bf16_kernel<<<dim3(144, 4), 256, 0, stream>>>(Wv, WvT, 192, 192);
  transpose_bf16_kernel<<<dim3(2304, 1), 256, 0, stream>>>(rk, rkT, DM, DM);
  transpose_bf16_kernel<<<dim3(2304, 1), 256, 0, stream>>>(Wo, WoT, DM, DM);

  // 3 projections * 128 M-tiles * 48 N-tiles = 18432 waves / 8 per block
  proj_qkv_kernel<<<2304, 256, 0, stream>>>(qbf, WqT, WkT, bk, WvT, bv, rwb,
                                            rrb, qu, qv, kbuf, vT);
  // 128 * 48 = 6144 waves / 8 per block
  rhead_kernel<<<768, 256, 0, stream>>>(pebf, rkT, rh);
  // B * NH * 64 query tiles = 1536 single-wave blocks (64KB LDS each)
  attn_kernel<<<1536, 32, 0, stream>>>(qu, qv, kbuf, vT, rh, av);
  out_proj_kernel<<<768, 256, 0, stream>>>(av, WoT, bo, qin, h);
  ln_kernel<<<2048, 256, 0, stream>>>(h, gmm, bta, (float*)d_out);
}
